// GSE3Res_20031727468996
// MI455X (gfx1250) — matrix operand order repaired and checked by reading, not compiled, not run on hardware
//
#include <hip/hip_runtime.h>

#define NN 4096
#define EE 32768
#define HH 4

typedef __attribute__((ext_vector_type(16))) _Float16 v16h;
typedef __attribute__((ext_vector_type(8)))  _Float16 v8h;
typedef __attribute__((ext_vector_type(8)))  float    v8f;

// ---------------------------------------------------------------------------
// Flat param offsets (floats) inside d_in[8], assuming jax pytree flattening
// (dict keys sorted at every level):
//   top: k, p0, p1, q0, q1, v
//   each radial net block (sorted): b1,b2,b3,bn1,bn2,g1,g2,w1,w2,w3
//   net block size = 2240 + 33*dout
// Net order used everywhere below: [v00, v01, v10, v11, k00, k01, k10, k11]
// ---------------------------------------------------------------------------
__constant__ int c_pbase[8] = {34112, 44800, 51264, 57728, 0, 10688, 17152, 23616};
__constant__ int c_dout [8] = {256, 128, 128, 192, 256, 128, 128, 192};
// pre-swizzled f16 B-fragment offsets (in halves) inside wsw region
__constant__ int c_wsoff[8] = {0, 10240, 16384, 24576, 30720, 40960, 47104, 55296};

#define P_P0 32192
#define P_P1 33472
#define P_Q0 33792
#define P_Q1 34048

// ---------------------------------------------------------------------------
// WMMA helpers
// ---------------------------------------------------------------------------
static __device__ inline v8f wmma16(v16h a, v16h b, v8f c) {
  return __builtin_amdgcn_wmma_f32_16x16x32_f16(false, a, false, b, (short)0, c,
                                                false, false);
}

// A-fragment 16x32 f16: lane holds row m = lane&15; hi = lane>>4 selects K sets
// halves 0..7 -> k = 8*hi + h ; halves 8..15 -> k = 16 + 8*hi + (h-8)
static __device__ inline v16h load_a(const _Float16* y, int m, int hi) {
  v8h lo = *(const v8h*)(y + m * 32 + 8 * hi);
  v8h hh = *(const v8h*)(y + m * 32 + 16 + 8 * hi);
  v16h a;
#pragma unroll
  for (int i = 0; i < 8; ++i) { a[i] = lo[i]; a[i + 8] = hh[i]; }
  return a;
}

// B-fragment 32x16 f16 pre-swizzled in memory: 16 halves per lane, contiguous.
static __device__ inline v16h load_b(const _Float16* p) {
  v8h lo = *(const v8h*)(p);
  v8h hh = *(const v8h*)(p + 8);
  v16h b;
#pragma unroll
  for (int i = 0; i < 8; ++i) { b[i] = lo[i]; b[i + 8] = hh[i]; }
  return b;
}

// xor-butterfly within 16-lane rows on the VALU pipe (v_permlane16_b32)
// instead of ds_bpermute: lanesel nibble j = j^mask.
static __device__ inline float plx(float x, int s0, int s1) {
  int v = __builtin_amdgcn_permlane16(__float_as_int(x), __float_as_int(x),
                                      s0, s1, false, false);
  return __int_as_float(v);
}
#define XOR1_LO  0x67452301
#define XOR1_HI  0xEFCDAB89
#define XOR2_LO  0x54761032
#define XOR2_HI  0xDCFE98BA
#define XOR4_LO  0x32107654
#define XOR4_HI  0xBA98FEDC
#define XOR8_LO  0xFEDCBA98
#define XOR8_HI  0x76543210

// LayerNorm(+ReLU) over 32 channels, directly on the two C fragments.
// C layout: VGPR v holds row m = hi*8+v, lane column n = lane&15 (c0: n, c1: n+16)
static __device__ inline void ln_relu(v8f& c0, v8f& c1, float g0, float g1,
                                      float b0, float b1) {
#pragma unroll
  for (int v = 0; v < 8; ++v) {
    float x0 = c0[v], x1 = c1[v];
    float s = x0 + x1, q = x0 * x0 + x1 * x1;
    s += plx(s, XOR1_LO, XOR1_HI);  q += plx(q, XOR1_LO, XOR1_HI);
    s += plx(s, XOR2_LO, XOR2_HI);  q += plx(q, XOR2_LO, XOR2_HI);
    s += plx(s, XOR4_LO, XOR4_HI);  q += plx(q, XOR4_LO, XOR4_HI);
    s += plx(s, XOR8_LO, XOR8_HI);  q += plx(q, XOR8_LO, XOR8_HI);
    float mu  = s * 0.03125f;
    float var = q * 0.03125f - mu * mu;
    float inv = rsqrtf(var + 1e-5f);
    c0[v] = fmaxf((x0 - mu) * inv * g0 + b0, 0.f);
    c1[v] = fmaxf((x1 - mu) * inv * g1 + b1, 0.f);
  }
}

// ---------------------------------------------------------------------------
// Weight prep: convert f32 params -> pre-swizzled f16 B fragments.
// Fragment layout: frag*512 + lane*16 + h ; element = W[n, k] with
// n = ntile*16 + (lane&15), k = 8*(lane>>4) + (h&7) + 16*(h>>3).
// Per net: frags 0-1 = layer1 (w1, k<32), 2-3 = layer2 (w2), 4.. = layer3 (w3)
// ---------------------------------------------------------------------------
__global__ void prep_weights(const float* __restrict__ P, _Float16* __restrict__ wsw) {
  const int net   = blockIdx.x;
  const int pbase = c_pbase[net];
  const int dout  = c_dout[net];
  const int off   = c_wsoff[net];
  const int nfr   = 4 + (dout >> 4);
  const int total = nfr * 512;
  const float* w1 = P + pbase + 192 + dout;
  const float* w2 = w1 + 1056;
  const float* w3 = w2 + 1024;
  for (int idx = threadIdx.x; idx < total; idx += blockDim.x) {
    int frag = idx >> 9;
    int rem  = idx & 511;
    int lane = rem >> 4;
    int h    = rem & 15;
    int nlo  = lane & 15;
    int k    = ((lane >> 4) << 3) + (h & 7) + ((h >> 3) << 4);
    float val;
    if (frag < 2)       val = w1[(frag * 16 + nlo) * 33 + k];
    else if (frag < 4)  val = w2[((frag - 2) * 16 + nlo) * 32 + k];
    else                val = w3[((frag - 4) * 16 + nlo) * 32 + k];
    wsw[off + idx] = (_Float16)val;
  }
}

// ---------------------------------------------------------------------------
// Heavy kernel: radial MLPs (WMMA) + basis contraction -> per-edge v/k messages
// vk layout per edge: [v0(8), v1(12), k0(8), k1(12)] = 40 floats
// ---------------------------------------------------------------------------
__global__ void __launch_bounds__(128) edge_kernel(
    const float* __restrict__ h0, const float* __restrict__ h1,
    const float* __restrict__ r,  const float* __restrict__ w,
    const float* __restrict__ b00, const float* __restrict__ b01,
    const float* __restrict__ b10, const float* __restrict__ b11,
    const float* __restrict__ P,  const int* __restrict__ src,
    const _Float16* __restrict__ wsw, float* __restrict__ vk) {
  struct WaveLds {
    _Float16 yA[512];     // 16x32 f16 A staging (feat / y1 / y2), row-major
    _Float16 Rl[4096];    // layer-3 tile, COLUMN-major: [col<=255][edge 0..15]
    float    rt[16];      // r values for the tile
  };
  __shared__ __align__(16) WaveLds S[4];
  const int wv   = threadIdx.x >> 5;
  const int lane = threadIdx.x & 31;
  const int nlo  = lane & 15;
  const int hi   = lane >> 4;
  WaveLds& s = S[wv];
  const int e0 = (blockIdx.x * 4 + wv) * 16;

  // stage feat tile (w channels) as f16, coalesced
#pragma unroll
  for (int i = 0; i < 16; ++i) {
    int idx = lane + 32 * i;
    int el = idx >> 5, k = idx & 31;
    s.yA[el * 32 + k] = (_Float16)w[(size_t)(e0 + el) * 32 + k];
  }
  if (lane < 16) s.rt[lane] = r[e0 + lane];
  __syncthreads();

  v16h afeat = load_a(s.yA, nlo, hi);
  float rv[8];
#pragma unroll
  for (int v = 0; v < 8; ++v) rv[v] = s.rt[hi * 8 + v];

  const int e  = e0 + nlo;
  const int sn = src[e];
  const float* h0row = h0 + (size_t)sn * 32;
  const float* h1row = h1 + (size_t)sn * 48;

  // R element (edge = nlo, flat column col) lives at Rl[col*16 + nlo]
#define RLD(col) ((float)s.Rl[(col) * 16 + nlo])

  for (int br = 0; br < 2; ++br) {          // 0 = 'v' branch, 1 = 'k' branch
    float m0[4] = {0.f, 0.f, 0.f, 0.f};
    float m1[6] = {0.f, 0.f, 0.f, 0.f, 0.f, 0.f};
    for (int nid = 0; nid < 4; ++nid) {     // (di,do) = 00, 01, 10, 11
      const int net  = br * 4 + nid;
      const int dout = c_dout[net];
      const float* pb = P + c_pbase[net];
      const _Float16* wf = wsw + c_wsoff[net];

      // ---- layer 1: feat @ w1^T (K=32 WMMA + rank-1 r column) + LN + ReLU
      v8f c0 = {}, c1 = {};
      {
        v16h bb0 = load_b(wf + lane * 16);
        v16h bb1 = load_b(wf + 512 + lane * 16);
        c0 = wmma16(afeat, bb0, c0);
        c1 = wmma16(afeat, bb1, c1);
        const float* w1p = pb + 192 + dout;
        float bi0 = pb[nlo], bi1 = pb[16 + nlo];
        float wr0 = w1p[nlo * 33 + 32], wr1 = w1p[(16 + nlo) * 33 + 32];
#pragma unroll
        for (int v = 0; v < 8; ++v) {
          c0[v] += bi0 + rv[v] * wr0;
          c1[v] += bi1 + rv[v] * wr1;
        }
        float g0 = pb[128 + dout + nlo], g1 = pb[128 + dout + 16 + nlo];
        float n0 = pb[64 + dout + nlo],  n1 = pb[64 + dout + 16 + nlo];
        ln_relu(c0, c1, g0, g1, n0, n1);
      }
#pragma unroll
      for (int v = 0; v < 8; ++v) {
        s.yA[(hi * 8 + v) * 32 + nlo]      = (_Float16)c0[v];
        s.yA[(hi * 8 + v) * 32 + 16 + nlo] = (_Float16)c1[v];
      }
      __syncthreads();
      v16h a1 = load_a(s.yA, nlo, hi);

      // ---- layer 2
      {
        v8f d0 = {}, d1 = {};
        v16h bb0 = load_b(wf + 1024 + lane * 16);
        v16h bb1 = load_b(wf + 1536 + lane * 16);
        d0 = wmma16(a1, bb0, d0);
        d1 = wmma16(a1, bb1, d1);
        float bi0 = pb[32 + nlo], bi1 = pb[48 + nlo];
#pragma unroll
        for (int v = 0; v < 8; ++v) { d0[v] += bi0; d1[v] += bi1; }
        float g0 = pb[160 + dout + nlo], g1 = pb[160 + dout + 16 + nlo];
        float n0 = pb[96 + dout + nlo],  n1 = pb[96 + dout + 16 + nlo];
        ln_relu(d0, d1, g0, g1, n0, n1);
#pragma unroll
        for (int v = 0; v < 8; ++v) {
          s.yA[(hi * 8 + v) * 32 + nlo]      = (_Float16)d0[v];
          s.yA[(hi * 8 + v) * 32 + 16 + nlo] = (_Float16)d1[v];
        }
      }
      __syncthreads();
      v16h a2 = load_a(s.yA, nlo, hi);

      // ---- layer 3: dout/16 WMMAs; stage R column-major with one b128 store
      const int nfr = dout >> 4;
      const float* b3 = pb + 64;
      for (int t = 0; t < nfr; ++t) {
        v16h bbf = load_b(wf + 2048 + t * 512 + lane * 16);
        v8f cc = {};
        cc = wmma16(a2, bbf, cc);
        float bias = b3[t * 16 + nlo];
        v8h pk;
#pragma unroll
        for (int v = 0; v < 8; ++v) pk[v] = (_Float16)(cc[v] + bias);
        // column = t*16+nlo, rows hi*8 .. hi*8+7 contiguous -> 16B store
        *(v8h*)(s.Rl + (t * 16 + nlo) * 16 + hi * 8) = pk;
      }
      __syncthreads();

      // ---- basis contraction (lane: edge = nlo, co-half = hi)
      if (nid == 0) {                       // (0,0): msg0 += b00 * R(8x32) @ src0
        float a0 = 0, a1v = 0, a2v = 0, a3 = 0;
        const int cb = hi * 4 * 32;
        for (int ci = 0; ci < 32; ++ci) {
          float sv = h0row[ci];
          a0  += RLD(cb + ci) * sv;        a1v += RLD(cb + 32 + ci) * sv;
          a2v += RLD(cb + 64 + ci) * sv;   a3  += RLD(cb + 96 + ci) * sv;
        }
        float bb = b00[e];
        m0[0] += bb * a0; m0[1] += bb * a1v; m0[2] += bb * a2v; m0[3] += bb * a3;
      } else if (nid == 1) {                // (0,1): u = R(4x32)@src0 ; msg1 += u (x) b01
        float u0 = 0, u1 = 0;
        const int cb = hi * 2 * 32;
        for (int ci = 0; ci < 32; ++ci) {
          float sv = h0row[ci];
          u0 += RLD(cb + ci) * sv;
          u1 += RLD(cb + 32 + ci) * sv;
        }
#pragma unroll
        for (int j = 0; j < 3; ++j) {
          float bj = b01[e * 3 + j];
          m1[j]     += u0 * bj;
          m1[3 + j] += u1 * bj;
        }
      } else if (nid == 2) {                // (1,0): t1[i]=b10.src1[i]; msg0 += R(8x16)@t1
        float p0v = b10[e * 3], p1v = b10[e * 3 + 1], p2v = b10[e * 3 + 2];
        const int cb = hi * 4 * 16;
        for (int i = 0; i < 16; ++i) {
          float t1 = p0v * h1row[i * 3] + p1v * h1row[i * 3 + 1] + p2v * h1row[i * 3 + 2];
          m0[0] += RLD(cb + i) * t1;       m0[1] += RLD(cb + 16 + i) * t1;
          m0[2] += RLD(cb + 32 + i) * t1;  m0[3] += RLD(cb + 48 + i) * t1;
        }
      } else {                              // (1,1): msg1 += sum_{i,f} R * (b11 . src1)
        float bl[27];
#pragma unroll
        for (int q2 = 0; q2 < 27; ++q2) bl[q2] = b11[(size_t)e * 27 + q2];
        const int cb = hi * 2 * 48;
        for (int i = 0; i < 16; ++i) {
          float s0v = h1row[i * 3], s1v = h1row[i * 3 + 1], s2v = h1row[i * 3 + 2];
#pragma unroll
          for (int j = 0; j < 3; ++j) {
            float T0 = bl[j * 9 + 0] * s0v + bl[j * 9 + 3] * s1v + bl[j * 9 + 6] * s2v;
            float T1 = bl[j * 9 + 1] * s0v + bl[j * 9 + 4] * s1v + bl[j * 9 + 7] * s2v;
            float T2 = bl[j * 9 + 2] * s0v + bl[j * 9 + 5] * s1v + bl[j * 9 + 8] * s2v;
#pragma unroll
            for (int c2 = 0; c2 < 2; ++c2) {
              const int rb = cb + c2 * 48 + i * 3;
              m1[c2 * 3 + j] += RLD(rb) * T0 + RLD(rb + 1) * T1 + RLD(rb + 2) * T2;
            }
          }
        }
      }
      __syncthreads();
    }
    // write this branch's messages
    float* o = vk + (size_t)e * 40 + br * 20;
#pragma unroll
    for (int c2 = 0; c2 < 4; ++c2) o[hi * 4 + c2] = m0[c2];
#pragma unroll
    for (int c2 = 0; c2 < 2; ++c2)
#pragma unroll
      for (int j = 0; j < 3; ++j) o[8 + (hi * 2 + c2) * 3 + j] = m1[c2 * 3 + j];
  }
#undef RLD
}

// ---------------------------------------------------------------------------
// Attention + output kernels
// ---------------------------------------------------------------------------
static __device__ inline unsigned ford(float f) {
  unsigned u = __float_as_uint(f);
  return (u & 0x80000000u) ? ~u : (u | 0x80000000u);
}
static __device__ inline float forddec(unsigned u) {
  unsigned b = (u & 0x80000000u) ? (u & 0x7fffffffu) : ~u;
  return __uint_as_float(b);
}

__global__ void init_kernel(float* __restrict__ z, float* __restrict__ esum,
                            unsigned* __restrict__ emaxb) {
  int t = blockIdx.x * blockDim.x + threadIdx.x;
  if (t < NN * 20)            z[t] = 0.f;
  else if (t < NN * 24)       esum[t - NN * 20] = 0.f;
  else if (t < NN * 28)       emaxb[t - NN * 24] = 0u;   // ordered -inf
}

__global__ void qk_kernel(const float* __restrict__ h0, const float* __restrict__ h1,
                          const float* __restrict__ P, float* __restrict__ qh) {
  int t = blockIdx.x * blockDim.x + threadIdx.x;
  if (t >= NN * HH) return;
  int n = t >> 2, h = t & 3;
  const float* q0w = P + P_Q0;
  const float* q1w = P + P_Q1;
  const float* h0r = h0 + (size_t)n * 32;
  const float* h1r = h1 + (size_t)n * 48;
  float* o = qh + n * 20 + h * 5;
  float a0 = 0, a1 = 0;
  for (int c = 0; c < 32; ++c) {
    a0 += q0w[(2 * h) * 32 + c] * h0r[c];
    a1 += q0w[(2 * h + 1) * 32 + c] * h0r[c];
  }
  o[0] = a0; o[1] = a1;
#pragma unroll
  for (int j = 0; j < 3; ++j) {
    float a = 0;
    for (int c = 0; c < 16; ++c) a += q1w[h * 16 + c] * h1r[c * 3 + j];
    o[2 + j] = a;
  }
}

__global__ void att1_kernel(const float* __restrict__ vk, const float* __restrict__ qh,
                            const int* __restrict__ dst, float* __restrict__ ebuf,
                            unsigned* __restrict__ emaxb) {
  int t = blockIdx.x * blockDim.x + threadIdx.x;
  if (t >= EE * HH) return;
  int e = t >> 2, h = t & 3;
  int n = dst[e];
  const float* kk = vk + (size_t)e * 40 + 20;
  const float* q  = qh + (size_t)n * 20 + h * 5;
  float ev = kk[2 * h] * q[0] + kk[2 * h + 1] * q[1] +
             kk[8 + h * 3] * q[2] + kk[8 + h * 3 + 1] * q[3] + kk[8 + h * 3 + 2] * q[4];
  ev *= 0.22360679775f;   // 1/sqrt(20)
  ebuf[t] = ev;
  atomicMax(emaxb + n * 4 + h, ford(ev));
}

__global__ void att2_kernel(const float* __restrict__ ebuf, const int* __restrict__ dst,
                            const unsigned* __restrict__ emaxb, float* __restrict__ exbuf,
                            float* __restrict__ esum) {
  int t = blockIdx.x * blockDim.x + threadIdx.x;
  if (t >= EE * HH) return;
  int e = t >> 2, h = t & 3;
  int n = dst[e];
  float ex = __expf(ebuf[t] - forddec(emaxb[n * 4 + h]));
  exbuf[t] = ex;
  atomicAdd(esum + n * 4 + h, ex);
}

__global__ void att3_kernel(const float* __restrict__ exbuf, const int* __restrict__ dst,
                            const float* __restrict__ esum, const float* __restrict__ vk,
                            float* __restrict__ z) {
  int t = blockIdx.x * blockDim.x + threadIdx.x;
  if (t >= EE * HH) return;
  int e = t >> 2, h = t & 3;
  int n = dst[e];
  float a = exbuf[t] / esum[n * 4 + h];
  const float* vv = vk + (size_t)e * 40;
  atomicAdd(z + n * 20 + 2 * h,     a * vv[2 * h]);
  atomicAdd(z + n * 20 + 2 * h + 1, a * vv[2 * h + 1]);
#pragma unroll
  for (int j = 0; j < 3; ++j)
    atomicAdd(z + n * 20 + 8 + h * 3 + j, a * vv[8 + h * 3 + j]);
}

__global__ void out_kernel(const float* __restrict__ h0, const float* __restrict__ h1,
                           const float* __restrict__ P, const float* __restrict__ z,
                           float* __restrict__ out) {
  int t = blockIdx.x * blockDim.x + threadIdx.x;
  if (t < NN * 32) {
    int n = t >> 5, o = t & 31;
    const float* p0 = P + P_P0 + o * 40;
    const float* zr = z + (size_t)n * 20;
    const float* hr = h0 + (size_t)n * 32;
    float acc = 0.f;
#pragma unroll
    for (int c = 0; c < 8; ++c) acc += p0[c] * zr[c];
#pragma unroll
    for (int c = 0; c < 32; ++c) acc += p0[8 + c] * hr[c];
    out[t] = acc;
  } else if (t < NN * 80) {
    int t2 = t - NN * 32;
    int n = t2 / 48; int rem = t2 - n * 48; int o = rem / 3; int j = rem - o * 3;
    const float* p1 = P + P_P1 + o * 20;
    const float* zr = z + (size_t)n * 20 + 8;
    const float* hr = h1 + (size_t)n * 48;
    float acc = 0.f;
#pragma unroll
    for (int c = 0; c < 4; ++c) acc += p1[c] * zr[c * 3 + j];
#pragma unroll
    for (int c = 0; c < 16; ++c) acc += p1[4 + c] * hr[c * 3 + j];
    out[NN * 32 + t2] = acc;
  }
}

// ---------------------------------------------------------------------------
// Launch
// ---------------------------------------------------------------------------
extern "C" void kernel_launch(void* const* d_in, const int* in_sizes, int n_in,
                              void* d_out, int out_size, void* d_ws, size_t ws_size,
                              hipStream_t stream) {
  (void)in_sizes; (void)n_in; (void)out_size; (void)ws_size;
  const float* h0  = (const float*)d_in[0];
  const float* h1  = (const float*)d_in[1];
  const float* r   = (const float*)d_in[2];
  const float* w   = (const float*)d_in[3];
  const float* b00 = (const float*)d_in[4];
  const float* b01 = (const float*)d_in[5];
  const float* b10 = (const float*)d_in[6];
  const float* b11 = (const float*)d_in[7];
  const float* P   = (const float*)d_in[8];
  const int* src   = (const int*)d_in[9];
  const int* dst   = (const int*)d_in[10];

  char* wsb = (char*)d_ws;
  _Float16* wsw   = (_Float16*)(wsb);            // 61440 halves = 122880 B
  float* vk       = (float*)(wsb + 122880);      // E*40 f32 = 5242880 B
  float* qh       = (float*)(wsb + 5365760);     // N*20
  float* ebuf     = (float*)(wsb + 5693440);     // E*4
  float* exbuf    = (float*)(wsb + 6217728);     // E*4
  unsigned* emaxb = (unsigned*)(wsb + 6742016);  // N*4
  float* esum     = (float*)(wsb + 6807552);     // N*4
  float* z        = (float*)(wsb + 6873088);     // N*20
  float* out      = (float*)d_out;

  prep_weights<<<8, 256, 0, stream>>>(P, wsw);
  init_kernel<<<(NN * 28 + 255) / 256, 256, 0, stream>>>(z, esum, emaxb);
  edge_kernel<<<EE / 64, 128, 0, stream>>>(h0, h1, r, w, b00, b01, b10, b11, P, src, wsw, vk);
  qk_kernel<<<(NN * HH + 255) / 256, 256, 0, stream>>>(h0, h1, P, qh);
  att1_kernel<<<(EE * HH) / 256, 256, 0, stream>>>(vk, qh, dst, ebuf, emaxb);
  att2_kernel<<<(EE * HH) / 256, 256, 0, stream>>>(ebuf, dst, emaxb, exbuf, esum);
  att3_kernel<<<(EE * HH) / 256, 256, 0, stream>>>(exbuf, dst, esum, vk, z);
  out_kernel<<<(NN * 80 + 255) / 256, 256, 0, stream>>>(h0, h1, P, z, out);
}
